// EvoformerBlock_55688545960638
// MI455X (gfx1250) — compile-verified, matrix operand order checked by reading, not run on hardware
//
#include <hip/hip_runtime.h>

// ============================================================================
// Evoformer block for MI455X (gfx1250, wave32, WMMA).
// All heavy math runs through v_wmma_f32_16x16x32_f16 (f32->f16 staged in LDS,
// f32 accumulate).  Attention is fused per (l,h) with the full 192x192 score
// tile resident in LDS (258KB of the 320KB/WGP CDNA5 budget).
// GEMM: templated fast paths (128b global loads), double-buffered LDS with
// software pipelining (fetch k+1 regs -> WMMA k -> store k+1 -> 1 barrier),
// division-free inner loop.  Attention: templated bias mode, compile-time
// S=192/H=8/d=32 geometry, 32-bit addressing.
// Workspace requirement: ~61,636,608 floats (~247 MB) in d_ws.
// ============================================================================

typedef __attribute__((ext_vector_type(16))) _Float16 v16h;
typedef __attribute__((ext_vector_type(8)))  _Float16 v8h;
typedef __attribute__((ext_vector_type(4)))  _Float16 v4h;
typedef __attribute__((ext_vector_type(8)))  float    v8f;

#define ACT_NONE 0
#define ACT_RELU 1
#define ACT_SIG  2

// Load one 16x32-f16 A/B fragment row per lane, per ISA 7.12.2:
// lane m = lane&15; halves: K = {lh*8..lh*8+7, 16+lh*8..+7}.  Two 16B LDS loads.
__device__ __forceinline__ v16h load_frag(const _Float16* __restrict__ p, int lh) {
  v8h lo = *(const v8h*)(p + lh * 8);
  v8h hi = *(const v8h*)(p + 16 + lh * 8);
  v16h r;
#pragma unroll
  for (int i = 0; i < 8; ++i) { r[i] = lo[i]; r[8 + i] = hi[i]; }
  return r;
}

// ----------------------------------------------------------------------------
// Generic WMMA GEMM: D[m,n] = epilogue( scale * sum_k A[m,k]*B[k,n] + bias[n] )
// A address: (m/m_lo)*sa_hi + (m%m_lo)*sa_lo + (k/k_lo)*sk_hi + (k%k_lo)*sk_lo
// B address: k*sb_k + n*sb_n
// D address: (m/d_mlo)*sd_hi + (m%d_mlo)*sd_lo + n*sd_n
// Requirements: K % 32 == 0, k_lo % 32 == 0 (holds for every call site).
// Tile 64x64, 8 waves, K-step 32 -> one v_wmma_f32_16x16x32_f16 per tile/step.
// FASTA: A is k-contiguous (sk_lo==1) -> float4 loads.
// FASTB: B is n-contiguous (sb_n==1)  -> float4 loads (transposed LDS store).
// ----------------------------------------------------------------------------
template <bool FASTA, bool FASTB>
__global__ __launch_bounds__(256) void gemm_wmma_kernel(
    const float* __restrict__ A, const float* __restrict__ Bw,
    const float* __restrict__ bias, const float* __restrict__ resid,
    const float* __restrict__ gate, float* __restrict__ D,
    int M, int N, int K,
    int m_lo, long sa_hi, long sa_lo,
    int k_lo, long sk_hi, long sk_lo, long a_bs,
    long sb_k, long sb_n, long b_bs,
    int d_mlo, long sd_hi, long sd_lo, long sd_n, long d_bs,
    int act, float scale)
{
  __shared__ _Float16 As[2][64][32];  // double-buffered A tile [m][k]
  __shared__ _Float16 Bs[2][64][32];  // double-buffered B tile, transposed [n][k]

  const int  bm    = blockIdx.x * 64;
  const int  bn    = blockIdx.y * 64;
  const long batch = blockIdx.z;
  const float* Ab = A + batch * a_bs;
  const float* Bb = Bw + batch * b_bs;
  float*       Db = D + batch * d_bs;
  const float* Rb = resid ? resid + batch * d_bs : nullptr;

  const int tid  = threadIdx.x;
  const int lane = tid & 31;
  const int wave = tid >> 5;        // 8 waves
  const int tr   = wave >> 1;       // wave's 16-row tile (0..3)
  const int tc0  = (wave & 1) * 2;  // first of two 16-col tiles
  const int lh   = lane >> 4;
  const int ll   = lane & 15;

  // --- per-thread staging slots; all m/n address math hoisted out of the loop.
  int  ar[2], ac4[2]; bool av[2]; long abase[2];
#pragma unroll
  for (int j = 0; j < 2; ++j) {
    int qidx = tid + 256 * j;
    ar[j]  = qidx >> 3;
    ac4[j] = (qidx & 7) * 4;
    int gm = bm + ar[j];
    av[j] = (gm < M);
    int gmc = av[j] ? gm : 0;
    abase[j] = (long)(gmc / m_lo) * sa_hi + (long)(gmc % m_lo) * sa_lo;
  }
  int  bkk[2], bn4[2]; bool bvv[2]; long bkb[2], bnb[2];
#pragma unroll
  for (int j = 0; j < 2; ++j) {
    int qidx = tid + 256 * j;
    bkk[j] = qidx >> 4;
    bn4[j] = (qidx & 15) * 4;
    int gn = bn + bn4[j];
    bvv[j] = (gn < N);
    bnb[j] = (long)(bvv[j] ? gn : 0) * sb_n;
    bkb[j] = (long)bkk[j] * sb_k;
  }

  float ax[2][4], bx[2][4];       // in-flight tile registers
  int khi = 0, klo = 0;           // block-uniform k decomposition counters

  auto fetch = [&]() {
    const long kbase = (long)khi * sk_hi + (long)klo * sk_lo;
#pragma unroll
    for (int j = 0; j < 2; ++j) {
      ax[j][0] = ax[j][1] = ax[j][2] = ax[j][3] = 0.f;
      if (av[j]) {
        const float* p = Ab + abase[j] + kbase;
        if (FASTA) {
          const float4 vv = *(const float4*)(p + ac4[j]);
          ax[j][0] = vv.x; ax[j][1] = vv.y; ax[j][2] = vv.z; ax[j][3] = vv.w;
        } else {
          long o = (long)ac4[j] * sk_lo;
          ax[j][0] = p[o];
          ax[j][1] = p[o + sk_lo];
          ax[j][2] = p[o + 2 * sk_lo];
          ax[j][3] = p[o + 3 * sk_lo];
        }
      }
    }
#pragma unroll
    for (int j = 0; j < 2; ++j) {
      bx[j][0] = bx[j][1] = bx[j][2] = bx[j][3] = 0.f;
      if (bvv[j]) {
        const float* p = Bb + bkb[j] + bnb[j];
        if (FASTB) {
          const float4 vv = *(const float4*)p;
          bx[j][0] = vv.x; bx[j][1] = vv.y; bx[j][2] = vv.z; bx[j][3] = vv.w;
        } else {
          bx[j][0] = p[0];
          bx[j][1] = p[sb_n];
          bx[j][2] = p[2 * sb_n];
          bx[j][3] = p[3 * sb_n];
        }
      }
      bkb[j] += 32 * sb_k;
    }
    klo += 32;
    if (klo >= k_lo) { klo = 0; ++khi; }
  };
  auto store_lds = [&](int buf) {
#pragma unroll
    for (int j = 0; j < 2; ++j) {
      v4h hv;
      hv[0] = (_Float16)ax[j][0]; hv[1] = (_Float16)ax[j][1];
      hv[2] = (_Float16)ax[j][2]; hv[3] = (_Float16)ax[j][3];
      *(v4h*)&As[buf][ar[j]][ac4[j]] = hv;
      Bs[buf][bn4[j] + 0][bkk[j]] = (_Float16)bx[j][0];
      Bs[buf][bn4[j] + 1][bkk[j]] = (_Float16)bx[j][1];
      Bs[buf][bn4[j] + 2][bkk[j]] = (_Float16)bx[j][2];
      Bs[buf][bn4[j] + 3][bkk[j]] = (_Float16)bx[j][3];
    }
  };

  v8f acc0 = {};
  v8f acc1 = {};

  // ---- software-pipelined main loop: one barrier per K-step.
  fetch();
  store_lds(0);
  __syncthreads();
  const int nk = K >> 5;
  for (int kk = 0; kk < nk; ++kk) {
    const int  cur  = kk & 1;
    const bool more = (kk + 1 < nk);
    if (more) {
      fetch();  // tile kk+1 -> registers (overlaps with WMMA below)
      __builtin_prefetch((const void*)(Ab + abase[0] +
                         (long)khi * sk_hi + (long)klo * sk_lo), 0, 1);
      __builtin_prefetch((const void*)(Bb + bkb[0] + bnb[0]), 0, 1);
    }
    v16h af = load_frag(&As[cur][tr * 16 + ll][0], lh);
    v16h b0 = load_frag(&Bs[cur][tc0 * 16 + ll][0], lh);
    v16h b1 = load_frag(&Bs[cur][(tc0 + 1) * 16 + ll][0], lh);
    acc0 = __builtin_amdgcn_wmma_f32_16x16x32_f16(false, af, false, b0,
                                                  (short)0, acc0, false, false);
    acc1 = __builtin_amdgcn_wmma_f32_16x16x32_f16(false, af, false, b1,
                                                  (short)0, acc1, false, false);
    if (more) store_lds(cur ^ 1);
    __syncthreads();
  }

  // ---- epilogue. C/D layout: lane n = ll (+tile), row m = r + 8*lh (+tile).
  const bool ddiv = (d_mlo < M);
  const int  gm0  = bm + tr * 16 + lh * 8;
  for (int t = 0; t < 2; ++t) {
    v8f acc = t ? acc1 : acc0;
    int gn = bn + (tc0 + t) * 16 + ll;
    if (gn >= N) continue;
    float bv = bias ? bias[gn] : 0.f;
    const long drow = (long)gm0 * sd_lo + (long)gn * sd_n;  // used when !ddiv
    const long grow = (long)gm0 * (long)N + gn;
#pragma unroll
    for (int r = 0; r < 8; ++r) {
      int gm = gm0 + r;
      if (gm >= M) continue;
      float v = acc[r] * scale + bv;
      if (act == ACT_RELU)      v = fmaxf(v, 0.f);
      else if (act == ACT_SIG)  v = 1.f / (1.f + __expf(-v));
      if (gate) v *= gate[grow + (long)r * N];
      long daddr = ddiv
          ? (long)(gm / d_mlo) * sd_hi + (long)(gm % d_mlo) * sd_lo + (long)gn * sd_n
          : (drow + (long)r * sd_lo);
      if (Rb) v += Rb[daddr];
      Db[daddr] = v;
    }
  }
}

// ----------------------------------------------------------------------------
// Fused attention for one (outer, head): S=192 keys/queries, head dim 32 —
// all geometry compile-time so every short loop unrolls and tile index math
// is shift/mul-by-constant (no runtime division).
// LDS: Qs/Ks [S][32] f16, Vt [32][S] f16, scores Sf [S][S] f32, P [S][S] f16
//      = 258,048 bytes (needs CDNA5's 320KB/WGP LDS).
// BMODE: 0 = no bias, 1 = full bias[.., s*b_s + t*b_t + h*b_h],
//        2 = query-row bias[.., s*b_s + h*b_h].
// ----------------------------------------------------------------------------
template <int BMODE>
__global__ __launch_bounds__(256) void attn_wmma_kernel(
    const float* __restrict__ Q, const float* __restrict__ Kx,
    const float* __restrict__ V, const float* __restrict__ Bias,
    float* __restrict__ O,
    long q_o, int q_s, int q_h,
    long k_o, int k_s, int k_h,
    long b_o, int b_s, int b_t, int b_h,
    long o_o, int o_s, int o_h,
    float scale)
{
  constexpr int S  = 192;
  constexpr int Hn = 8;
  constexpr int nt = S / 16;   // 12
  constexpr int per = S / 32;  // 6

  extern __shared__ char smem[];
  _Float16* Qs = (_Float16*)smem;          // [S][32]
  _Float16* Ks = Qs + S * 32;              // [S][32]
  _Float16* Vt = Ks + S * 32;              // [32][S] (transposed for B-frags)
  float*    Sf = (float*)(Vt + 32 * S);    // [S][S]
  _Float16* Pf = (_Float16*)(Sf + S * S);  // [S][S]

  const int outer = blockIdx.x / Hn;       // Hn constant -> shift
  const int h     = blockIdx.x % Hn;
  const int tid = threadIdx.x, lane = tid & 31, wave = tid >> 5;
  const int lh = lane >> 4, ll = lane & 15;

  const float* Qb = Q  + (long)outer * q_o + (long)h * q_h;
  const float* Kb = Kx + (long)outer * k_o + (long)h * k_h;
  const float* Vb = V  + (long)outer * k_o + (long)h * k_h;  // V shares K layout
  float*       Ob = O  + (long)outer * o_o + (long)h * o_h;
  const float* Bb = (BMODE != 0) ? (Bias + (long)outer * b_o + (long)h * b_h)
                                 : nullptr;

  // Vectorized staging: all layouts are d-contiguous and 16B aligned.
#pragma unroll
  for (int u = 0; u < (S * 8) / 256; ++u) {
    int i = tid + u * 256;
    int t = i >> 3, d4 = (i & 7) * 4;
    const float4 q4 = *(const float4*)(Qb + t * q_s + d4);
    const float4 k4 = *(const float4*)(Kb + t * k_s + d4);
    const float4 w4 = *(const float4*)(Vb + t * k_s + d4);
    v4h qh; qh[0]=(_Float16)q4.x; qh[1]=(_Float16)q4.y; qh[2]=(_Float16)q4.z; qh[3]=(_Float16)q4.w;
    v4h kh; kh[0]=(_Float16)k4.x; kh[1]=(_Float16)k4.y; kh[2]=(_Float16)k4.z; kh[3]=(_Float16)k4.w;
    *(v4h*)&Qs[t * 32 + d4] = qh;
    *(v4h*)&Ks[t * 32 + d4] = kh;
    Vt[(d4 + 0) * S + t] = (_Float16)w4.x;
    Vt[(d4 + 1) * S + t] = (_Float16)w4.y;
    Vt[(d4 + 2) * S + t] = (_Float16)w4.z;
    Vt[(d4 + 3) * S + t] = (_Float16)w4.w;
  }
  __syncthreads();

  // Scores: Sf[s,t] = scale * Q[s,:] . K[t,:] (+ bias).  144 tiles / 8 waves.
#pragma unroll 2
  for (int u = 0; u < (nt * nt) / 8; ++u) {
    int ti = wave + u * 8;
    int st = ti / nt, tt = ti % nt;      // nt constant -> strength-reduced
    v16h af = load_frag(Qs + (st * 16 + ll) * 32, lh);
    v16h bf = load_frag(Ks + (tt * 16 + ll) * 32, lh);
    v8f c = {};
    c = __builtin_amdgcn_wmma_f32_16x16x32_f16(false, af, false, bf,
                                               (short)0, c, false, false);
    const int n   = tt * 16 + ll;
    const int sr0 = st * 16 + lh * 8;
    const float* bpt = nullptr;
    if (BMODE == 1) bpt = Bb + sr0 * b_s + n * b_t;
    if (BMODE == 2) bpt = Bb + sr0 * b_s;
#pragma unroll
    for (int r = 0; r < 8; ++r) {
      float v = c[r] * scale;
      if (BMODE != 0) v += bpt[r * b_s];
      Sf[(sr0 + r) * S + n] = v;
    }
  }
  __syncthreads();

  // Row softmax, wave32 shuffle reduction (6 elements/lane at S=192).
#pragma unroll 2
  for (int row = wave; row < S; row += 8) {
    float* rp = Sf + row * S;
    float e[per];
    float mx = -3.0e38f;
#pragma unroll
    for (int j = 0; j < per; ++j) mx = fmaxf(mx, rp[lane + 32 * j]);
#pragma unroll
    for (int off = 16; off > 0; off >>= 1) mx = fmaxf(mx, __shfl_xor(mx, off, 32));
    float sum = 0.f;
#pragma unroll
    for (int j = 0; j < per; ++j) { e[j] = __expf(rp[lane + 32 * j] - mx); sum += e[j]; }
#pragma unroll
    for (int off = 16; off > 0; off >>= 1) sum += __shfl_xor(sum, off, 32);
    float inv = 1.f / sum;
#pragma unroll
    for (int j = 0; j < per; ++j)
      Pf[row * S + lane + 32 * j] = (_Float16)(e[j] * inv);
  }
  __syncthreads();

  // O[s,d] = P[s,:] . V[:,d]   (24 tiles / 8 waves, K = S in steps of 32)
#pragma unroll
  for (int u = 0; u < (nt * 2) / 8; ++u) {
    int ti = wave + u * 8;
    int st = ti >> 1, dt = ti & 1;
    v8f c = {};
#pragma unroll
    for (int kk = 0; kk < S; kk += 32) {
      v16h af = load_frag(Pf + (st * 16 + ll) * S + kk, lh);
      v16h bf = load_frag(Vt + (dt * 16 + ll) * S + kk, lh);
      c = __builtin_amdgcn_wmma_f32_16x16x32_f16(false, af, false, bf,
                                                 (short)0, c, false, false);
    }
    const int n   = dt * 16 + ll;
    const int sr0 = st * 16 + lh * 8;
#pragma unroll
    for (int r = 0; r < 8; ++r)
      Ob[(sr0 + r) * o_s + n] = c[r];
  }
}

// ----------------------------------------------------------------------------
// LayerNorm: one wave32 per row, C in {32,128,256}.
// ----------------------------------------------------------------------------
__global__ __launch_bounds__(256) void ln_kernel(
    const float* __restrict__ x, const float* __restrict__ g,
    const float* __restrict__ b, float* __restrict__ y, int C, long rows)
{
  const int lane = threadIdx.x & 31;
  const int wave = threadIdx.x >> 5;
  long row = (long)blockIdx.x * 8 + wave;
  if (row >= rows) return;
  const float* xr = x + row * C;
  float*       yr = y + row * C;
  const int per = C >> 5;
  float vals[8];
  float s = 0.f, ss = 0.f;
  for (int j = 0; j < per; ++j) {
    float v = xr[lane + 32 * j];
    vals[j] = v; s += v; ss += v * v;
  }
  for (int off = 16; off > 0; off >>= 1) {
    s  += __shfl_xor(s,  off, 32);
    ss += __shfl_xor(ss, off, 32);
  }
  float mu  = s / (float)C;
  float var = ss / (float)C - mu * mu;
  float inv = rsqrtf(var + 1e-5f);
  for (int j = 0; j < per; ++j) {
    int c = lane + 32 * j;
    yr[c] = (vals[j] - mu) * inv * g[c] + b[c];
  }
}

__global__ void mul_kernel(float* __restrict__ o, const float* __restrict__ g, long n) {
  long i = (long)blockIdx.x * blockDim.x + threadIdx.x;
  long st = (long)gridDim.x * blockDim.x;
  for (; i < n; i += st) o[i] *= g[i];
}

// ============================================================================
// Host orchestration
// ============================================================================
extern "C" void kernel_launch(void* const* d_in, const int* in_sizes, int n_in,
                              void* d_out, int out_size, void* d_ws, size_t ws_size,
                              hipStream_t stream) {
  (void)in_sizes; (void)n_in; (void)out_size; (void)ws_size;

  const int Lr = 192, Hn = 8;
  const long RM = 36864;               // NS*L == L*L rows
  const float ATT_SCALE = 0.17677669529663687f;  // 1/sqrt(32)

  auto F = [&](int i) { return (const float*)d_in[i]; };
  float* ws   = (float*)d_ws;
  float* mcur = (float*)d_out;           // m output region [192,192,256]
  float* zcur = (float*)d_out + 9437184; // z output region [192,192,128]
  const float* m_in = F(0);
  const float* z_in = F(1);

  // workspace arena (floats); peak usage ~61.6M floats (~247MB)
  const long off_mn = 0;                 // 9,437,184
  const long off_zn = 9437184;           // 4,718,592
  const long off_q  = 14155776;          // 9,437,184
  const long off_k  = 23592960;          // 9,437,184
  const long off_v  = 33030144;          // 9,437,184
  const long off_g  = 42467328;          // 9,437,184
  const long off_o  = 51904512;          // 9,437,184
  const long off_bb = 61341696;          //   294,912
  const long off_big = off_q;            // 37,748,736 (reuses q..g span)
  float* mn   = ws + off_mn;
  float* zn   = ws + off_zn;
  float* qB   = ws + off_q;
  float* kB   = ws + off_k;
  float* vB   = ws + off_v;
  float* gB   = ws + off_g;
  float* oB   = ws + off_o;
  float* bbB  = ws + off_bb;
  float* big  = ws + off_big;
  float* a2   = ws + off_o;              // 1,179,648 (trimul/opm a)
  float* b2   = ws + off_o + 1179648;    // 1,179,648 (trimul/opm b)
  float* abB  = ws + off_v;              // trimul ab
  float* abn  = ws + off_q;              // trimul LN(ab)
  float* gz   = ws + off_g;              // trimul gate

  const int attn_smem = (3 * 192 * 32) * 2 + 192 * 192 * 4 + 192 * 192 * 2; // 258048
  hipFuncSetAttribute((const void*)attn_wmma_kernel<0>,
                      hipFuncAttributeMaxDynamicSharedMemorySize, attn_smem);
  hipFuncSetAttribute((const void*)attn_wmma_kernel<1>,
                      hipFuncAttributeMaxDynamicSharedMemorySize, attn_smem);
  hipFuncSetAttribute((const void*)attn_wmma_kernel<2>,
                      hipFuncAttributeMaxDynamicSharedMemorySize, attn_smem);

  auto ln = [&](const float* x, const float* gg, const float* bb, float* y,
                int C, long rows) {
    ln_kernel<<<dim3((unsigned)((rows + 7) / 8)), 256, 0, stream>>>(x, gg, bb, y, C, rows);
  };
  auto gemm = [&](const float* A, const float* Bw, const float* bias,
                  const float* resid, const float* gate, float* D,
                  int M, int N, int K,
                  int m_lo, long sa_hi, long sa_lo,
                  int k_lo, long sk_hi, long sk_lo, long a_bs,
                  long sb_k, long sb_n, long b_bs,
                  int d_mlo, long sd_hi, long sd_lo, long sd_n, long d_bs,
                  int act, float scale, int batches) {
    dim3 grid((unsigned)((M + 63) / 64), (unsigned)((N + 63) / 64), (unsigned)batches);
    const bool fA = (sk_lo == 1), fB = (sb_n == 1);
    if (fA && fB)
      gemm_wmma_kernel<true, true><<<grid, 256, 0, stream>>>(
          A, Bw, bias, resid, gate, D, M, N, K,
          m_lo, sa_hi, sa_lo, k_lo, sk_hi, sk_lo, a_bs,
          sb_k, sb_n, b_bs, d_mlo, sd_hi, sd_lo, sd_n, d_bs, act, scale);
    else if (fB)
      gemm_wmma_kernel<false, true><<<grid, 256, 0, stream>>>(
          A, Bw, bias, resid, gate, D, M, N, K,
          m_lo, sa_hi, sa_lo, k_lo, sk_hi, sk_lo, a_bs,
          sb_k, sb_n, b_bs, d_mlo, sd_hi, sd_lo, sd_n, d_bs, act, scale);
    else
      gemm_wmma_kernel<false, false><<<grid, 256, 0, stream>>>(
          A, Bw, bias, resid, gate, D, M, N, K,
          m_lo, sa_hi, sa_lo, k_lo, sk_hi, sk_lo, a_bs,
          sb_k, sb_n, b_bs, d_mlo, sd_hi, sd_lo, sd_n, d_bs, act, scale);
  };
  auto gemm_std = [&](const float* A, const float* Bw, const float* bias,
                      const float* resid, const float* gate, float* D,
                      int M, int N, int K, int act) {
    gemm(A, Bw, bias, resid, gate, D, M, N, K,
         M, 0, (long)K, K, 0, 1, 0, (long)N, 1, 0,
         M, 0, (long)N, 1, 0, act, 1.0f, 1);
  };
  // transposed-row read of zn ([L,L,CZ] -> rows (i',j') read zn[j',i',:])
  auto gemm_zT = [&](const float* A, const float* Bw, const float* bias, float* D,
                     int N, int act) {
    gemm(A, Bw, bias, nullptr, nullptr, D, (int)RM, N, 128,
         192, 128, 24576, 128, 0, 1, 0, (long)N, 1, 0,
         (int)RM, 0, (long)N, 1, 0, act, 1.0f, 1);
  };
  auto attn = [&](int bmode,
                  const float* Q, const float* K, const float* V, const float* Bias,
                  float* O,
                  long q_o, int q_s, int q_h, long k_o, int k_s, int k_h,
                  long b_o, int b_s, int b_t, int b_h,
                  long o_o, int o_s, int o_h) {
    dim3 grid((unsigned)(Lr * Hn));
    if (bmode == 1)
      attn_wmma_kernel<1><<<grid, 256, attn_smem, stream>>>(
          Q, K, V, Bias, O, q_o, q_s, q_h, k_o, k_s, k_h,
          b_o, b_s, b_t, b_h, o_o, o_s, o_h, ATT_SCALE);
    else if (bmode == 2)
      attn_wmma_kernel<2><<<grid, 256, attn_smem, stream>>>(
          Q, K, V, Bias, O, q_o, q_s, q_h, k_o, k_s, k_h,
          b_o, b_s, b_t, b_h, o_o, o_s, o_h, ATT_SCALE);
    else
      attn_wmma_kernel<0><<<grid, 256, attn_smem, stream>>>(
          Q, K, V, Bias, O, q_o, q_s, q_h, k_o, k_s, k_h,
          b_o, b_s, b_t, b_h, o_o, o_s, o_h, ATT_SCALE);
  };
  auto emul = [&](float* o, const float* g, long n) {
    mul_kernel<<<dim3(4096), 256, 0, stream>>>(o, g, n);
  };

  // ---------------- row attention (attends over NS at fixed l, z-bias) -------
  ln(m_in, F(2), F(3), mn, 256, RM);
  ln(z_in, F(11), F(12), zn, 128, RM);
  gemm_std(zn, F(13), nullptr, nullptr, nullptr, bbB, (int)RM, 8, 128, ACT_NONE);
  gemm_std(mn, F(4), nullptr, nullptr, nullptr, qB, (int)RM, 256, 256, ACT_NONE);
  gemm_std(mn, F(5), nullptr, nullptr, nullptr, kB, (int)RM, 256, 256, ACT_NONE);
  gemm_std(mn, F(6), nullptr, nullptr, nullptr, vB, (int)RM, 256, 256, ACT_NONE);
  gemm_std(mn, F(7), F(8), nullptr, nullptr, gB, (int)RM, 256, 256, ACT_SIG);
  // q[s,l,h,d]: s-stride 49152, l-stride 256, h-stride 32; bias[s,t,h]
  attn(1, qB, kB, vB, bbB, oB,
       /*q*/ 256, 49152, 32, /*k*/ 256, 49152, 32,
       /*bias*/ 0, 1536, 8, 1, /*o*/ 256, 49152, 32);
  emul(oB, gB, 9437184);
  gemm_std(oB, F(9), F(10), m_in, nullptr, mcur, (int)RM, 256, 256, ACT_NONE);

  // ---------------- column attention (same math, quirky raw-reshape output) --
  ln(mcur, F(14), F(15), mn, 256, RM);
  gemm_std(mn, F(16), nullptr, nullptr, nullptr, qB, (int)RM, 256, 256, ACT_NONE);
  gemm_std(mn, F(17), nullptr, nullptr, nullptr, kB, (int)RM, 256, 256, ACT_NONE);
  gemm_std(mn, F(18), nullptr, nullptr, nullptr, vB, (int)RM, 256, 256, ACT_NONE);
  gemm_std(mn, F(19), F(20), nullptr, nullptr, gB, (int)RM, 256, 256, ACT_SIG);
  // faithful raw reshape: o_flat[((h*L + l)*S + s)*32 + d]
  attn(0, qB, kB, vB, nullptr, oB,
       256, 49152, 32, 256, 49152, 32,
       0, 0, 0, 0, /*o*/ 6144, 32, 1179648);
  emul(oB, gB, 9437184);
  gemm_std(oB, F(21), F(22), mcur, nullptr, mcur, (int)RM, 256, 256, ACT_NONE);

  // ---------------- MSA transition ------------------------------------------
  ln(mcur, F(23), F(24), mn, 256, RM);
  gemm_std(mn, F(25), F(26), nullptr, nullptr, big, (int)RM, 1024, 256, ACT_RELU);
  gemm_std(big, F(27), F(28), mcur, nullptr, mcur, (int)RM, 256, 1024, ACT_NONE);

  // ---------------- outer product mean --------------------------------------
  ln(mcur, F(29), F(30), mn, 256, RM);
  gemm_std(mn, F(31), F(32), nullptr, nullptr, a2, (int)RM, 32, 256, ACT_NONE);
  gemm_std(mn, F(33), F(34), nullptr, nullptr, b2, (int)RM, 32, 256, ACT_NONE);
  // outer[(i,c),(j,d)] = (1/NS) sum_s a[s,i,c] b[s,j,d]
  gemm(a2, b2, nullptr, nullptr, nullptr, big, 6144, 6144, 192,
       32, 32, 1, 192, 0, 6144, 0, 6144, 1, 0,
       6144, 0, 6144, 1, 0, ACT_NONE, 1.0f / 192.0f, 1);
  // z = z + vec(outer[i,j,:,:]) @ wo + bo  (A read in [(i,j),(c,d)] order)
  gemm(big, F(35), F(36), z_in, nullptr, zcur, (int)RM, 128, 1024,
       192, 196608, 32, 32, 6144, 1, 0, 128, 1, 0,
       (int)RM, 0, 128, 1, 0, ACT_NONE, 1.0f, 1);

  // ---------------- triangle multiply (outgoing then incoming) --------------
  for (int pass = 0; pass < 2; ++pass) {
    int p0 = 37 + pass * 12;            // tmo=37.., tmi=49..
    ln(zcur, F(p0 + 0), F(p0 + 1), zn, 128, RM);
    gemm_std(zn, F(p0 + 2), F(p0 + 3), nullptr, nullptr, a2, (int)RM, 32, 128, ACT_NONE);
    gemm_std(zn, F(p0 + 4), F(p0 + 5), nullptr, nullptr, b2, (int)RM, 32, 128, ACT_NONE);
    gemm_std(zn, F(p0 + 6), F(p0 + 7), nullptr, nullptr, gz, (int)RM, 128, 128, ACT_SIG);
    if (pass == 0) {
      // outgoing: AB_c[i,j] = sum_k a[i,k,c] b[j,k,c]   (32 channel batches)
      gemm(a2, b2, nullptr, nullptr, nullptr, abB, 192, 192, 192,
           192, 0, 6144, 192, 0, 32, 1, 32, 6144, 1,
           192, 0, 6144, 32, 1, ACT_NONE, 1.0f, 32);
    } else {
      // incoming: AB_c[i,j] = sum_k a[k,i,c] b[k,j,c]
      gemm(a2, b2, nullptr, nullptr, nullptr, abB, 192, 192, 192,
           192, 0, 32, 192, 0, 6144, 1, 6144, 32, 1,
           192, 0, 6144, 32, 1, ACT_NONE, 1.0f, 32);
    }
    ln(abB, F(p0 + 8), F(p0 + 9), abn, 32, RM);
    // z = z + g * (LN(ab) @ wo + bo)
    gemm_std(abn, F(p0 + 10), F(p0 + 11), zcur, gz, zcur, (int)RM, 128, 32, ACT_NONE);
  }

  // ---------------- triangle attention (starting then ending) ---------------
  for (int pass = 0; pass < 2; ++pass) {
    int p0 = 61 + pass * 10;            // tas=61.., tae=71..
    ln(zcur, F(p0 + 0), F(p0 + 1), zn, 128, RM);
    if (pass == 0) {
      gemm_std(zn, F(p0 + 2), nullptr, nullptr, nullptr, qB, (int)RM, 256, 128, ACT_NONE);
      gemm_std(zn, F(p0 + 3), nullptr, nullptr, nullptr, kB, (int)RM, 32, 128, ACT_NONE);
      gemm_std(zn, F(p0 + 4), nullptr, nullptr, nullptr, vB, (int)RM, 32, 128, ACT_NONE);
      gemm_std(zn, F(p0 + 5), nullptr, nullptr, nullptr, bbB, (int)RM, 8, 128, ACT_NONE);
      gemm_std(zn, F(p0 + 6), F(p0 + 7), nullptr, nullptr, gB, (int)RM, 256, 128, ACT_SIG);
    } else {
      // "ending": operate on transposed zn; projections read zn[j',i',:]
      gemm_zT(zn, F(p0 + 2), nullptr, qB, 256, ACT_NONE);
      gemm_zT(zn, F(p0 + 3), nullptr, kB, 32, ACT_NONE);
      gemm_zT(zn, F(p0 + 4), nullptr, vB, 32, ACT_NONE);
      gemm_zT(zn, F(p0 + 5), nullptr, bbB, 8, ACT_NONE);
      gemm_zT(zn, F(p0 + 6), F(p0 + 7), gB, 256, ACT_SIG);
    }
    // q[i,j,h,d]; single-head k,v[i,k,d]; bias bb[i,j,h] on the QUERY index
    attn(2, qB, kB, vB, bbB, oB,
         /*q*/ 49152, 256, 32, /*k*/ 6144, 32, 0,
         /*bias*/ 1536, 8, 0, 1, /*o*/ 49152, 256, 32);
    emul(oB, gB, 9437184);
    if (pass == 0) {
      gemm_std(oB, F(p0 + 8), F(p0 + 9), zcur, nullptr, zcur, (int)RM, 128, 256, ACT_NONE);
    } else {
      // store transposed back: z[j',i'] += o'[i',j'] @ wo + bo
      gemm(oB, F(p0 + 8), F(p0 + 9), zcur, nullptr, zcur, (int)RM, 128, 256,
           (int)RM, 0, 256, 256, 0, 1, 0, 128, 1, 0,
           192, 128, 24576, 1, 0, ACT_NONE, 1.0f, 1);
    }
  }

  // ---------------- pair transition -----------------------------------------
  ln(zcur, F(81), F(82), zn, 128, RM);
  gemm_std(zn, F(83), F(84), nullptr, nullptr, big, (int)RM, 512, 128, ACT_RELU);
  gemm_std(big, F(85), F(86), zcur, nullptr, zcur, (int)RM, 128, 512, ACT_NONE);
}
